// Predictor_47811575939714
// MI455X (gfx1250) — matrix operand order repaired and checked by reading, not compiled, not run on hardware
//
#include <hip/hip_runtime.h>
#include <math.h>

// ---------------------------------------------------------------------------
// CDNA5 (gfx1250, wave32) implementation.  All GEMMs run on
// v_wmma_f32_16x16x32_f16 (f16 inputs, f32 accumulate); softmax / layernorm /
// elementwise stay in f32.
//
// Round-2 codegen fixes vs round 1:
//  * branchless clamped staging loads (no exec-mask branches, no per-load waits)
//  * two-phase staging (batch loads -> one wait -> convert/store)
//  * B operand stored transposed+padded in LDS so fragment reads are
//    2x ds_load_b128 instead of 16x ds_load_u16
//  * K-step 64: two v_wmma per barrier pair
// ---------------------------------------------------------------------------

typedef __attribute__((ext_vector_type(16))) _Float16 v16h;
typedef __attribute__((ext_vector_type(8)))  _Float16 v8h;
typedef __attribute__((ext_vector_type(2)))  _Float16 v2h;
typedef __attribute__((ext_vector_type(8)))  float    v8f;

#define Bc    64
#define Sc    100
#define Tc    10
#define NNEGc 101
#define Rc    1010      /* NNEG*T */
#define DIMc  64
#define Dc    128
#define RADc  0.017453292519943295f

#define BT_LD 72        /* padded row stride (halves) of transposed B tile */

// ---- fragment loaders (CDNA5 wave32 layouts, ISA 7.12.2) -------------------
// A 16x32 f16 : lane m=lane&15; K-offsets {0..7,16..23} (lanes<16) or
//               {8..15,24..31} (lanes>=16).  Ah row-major, ld halves.
__device__ __forceinline__ v16h load_afrag(const _Float16* Ap, int ld,
                                           int lane, int kbase)
{
    const int mr = lane & 15;
    const int ko = (lane >> 4) << 3;
    const v8h a0 = *(const v8h*)(Ap + mr * ld + kbase + ko);
    const v8h a1 = *(const v8h*)(Ap + mr * ld + kbase + 16 + ko);
    return __builtin_shufflevector(a0, a1, 0, 1, 2, 3, 4, 5, 6, 7,
                                   8, 9, 10, 11, 12, 13, 14, 15);
}
// B 32x16 f16 : lane n=lane&15; K {0..7,16..23} (lanes<16), {8..15,24..31} (lanes>=16).
// Bt is TRANSPOSED in LDS: Bt[n * BT_LD + k]  -> contiguous per lane.
__device__ __forceinline__ v16h load_bfrag(const _Float16* Bt,
                                           int lane, int wid, int kbase)
{
    const int nc = (wid << 4) + (lane & 15);
    const int kb = (lane >> 4) << 3;
    const v8h b0 = *(const v8h*)(Bt + nc * BT_LD + kbase + kb);
    const v8h b1 = *(const v8h*)(Bt + nc * BT_LD + kbase + kb + 16);
    return __builtin_shufflevector(b0, b1, 0, 1, 2, 3, 4, 5, 6, 7,
                                   8, 9, 10, 11, 12, 13, 14, 15);
}

// ---------------------------------------------------------------------------
// Generic batched GEMM:  C[b] = scale * (A[b] @ B[b](ᵀ))
//   grid = (ceil(M/16), ceil(N/64), nbatch), block = 128 (4 waves)
//   K-step 64 -> 2 WMMAs per stage.  All K in this model are even.
// ---------------------------------------------------------------------------
__global__ void k_gemm(const float* __restrict__ A, long long bsA, int lda,
                       const float* __restrict__ Bm, long long bsB, int ldb, int bT,
                       float* __restrict__ C, long long bsC, int ldc,
                       int M, int N, int K, float scale)
{
    __shared__ alignas(16) _Float16 Ah[16 * 64];
    __shared__ alignas(16) _Float16 Bt[64 * BT_LD];

    const int tid  = threadIdx.x;
    const int lane = tid & 31;
    const int wid  = tid >> 5;
    const int m0   = blockIdx.x * 16;
    const int n0   = blockIdx.y * 64;
    const long long bz = blockIdx.z;
    const float* Ab = A  + bz * bsA;
    const float* Bb = Bm + bz * bsB;

    v8f acc = {};

    for (int k0 = 0; k0 < K; k0 += 64) {
        // ---- stage A (16 x 64) : branchless clamped float2 loads ----
        float2 av[4];
        #pragma unroll
        for (int i = 0; i < 4; ++i) {
            int g  = tid + 128 * i;           // 512 float2 groups
            int r  = g >> 5;
            int kg = (g & 31) << 1;
            int row = m0 + r; if (row > M - 1) row = M - 1;
            int kc  = k0 + kg; if (kc > K - 2) kc = K - 2;
            float2 v = *(const float2*)(Ab + (long long)row * lda + kc);
            bool ok = (m0 + r < M) && (k0 + kg < K);   // K even => pair valid
            av[i].x = ok ? v.x : 0.f;
            av[i].y = ok ? v.y : 0.f;
        }
        #pragma unroll
        for (int i = 0; i < 4; ++i) {
            int g  = tid + 128 * i;
            int r  = g >> 5;
            int kg = (g & 31) << 1;
            v2h h;
            h[0] = (_Float16)av[i].x;
            h[1] = (_Float16)av[i].y;
            *(v2h*)(Ah + r * 64 + kg) = h;
        }
        // ---- stage B (64 x 64) transposed into Bt[n][k], branchless ----
        #pragma unroll
        for (int bb = 0; bb < 4; ++bb) {
            float bv[8];
            #pragma unroll
            for (int i = 0; i < 8; ++i) {
                int e = tid + 128 * (bb * 8 + i);      // 0..4095
                int kk, nn;
                if (bT) { kk = e & 63; nn = e >> 6; }
                else    { kk = e >> 6; nn = e & 63; }
                int gk = k0 + kk; int gkc = gk > K - 1 ? K - 1 : gk;
                int gn = n0 + nn; int gnc = gn > N - 1 ? N - 1 : gn;
                float v = bT ? Bb[(long long)gnc * ldb + gkc]
                             : Bb[(long long)gkc * ldb + gnc];
                bv[i] = (gk < K && gn < N) ? v : 0.f;
            }
            #pragma unroll
            for (int i = 0; i < 8; ++i) {
                int e = tid + 128 * (bb * 8 + i);
                int kk, nn;
                if (bT) { kk = e & 63; nn = e >> 6; }
                else    { kk = e >> 6; nn = e & 63; }
                Bt[nn * BT_LD + kk] = (_Float16)bv[i];
            }
        }
        // prefetch next B chunk (global_prefetch_b8)
        if (k0 + 64 < K) {
            int pn = n0 < N - 1 ? n0 : N - 1;
            const float* nxt = bT ? &Bb[(long long)pn * ldb + k0 + 64]
                                  : &Bb[(long long)(k0 + 64) * ldb + pn];
            __builtin_prefetch(nxt, 0, 1);
        }
        __syncthreads();

        #pragma unroll
        for (int c = 0; c < 2; ++c) {
            v16h a = load_afrag(Ah, 64, lane, c * 32);
            v16h b = load_bfrag(Bt, lane, wid, c * 32);
            acc = __builtin_amdgcn_wmma_f32_16x16x32_f16(false, a, false, b,
                                                         (short)0, acc,
                                                         false, false);
        }
        __syncthreads();
    }

    float* Cb = C + bz * bsC;
    const int hi = (lane >> 4) * 8;
    const int nn = n0 + 16 * wid + (lane & 15);
    #pragma unroll
    for (int v = 0; v < 8; ++v) {
        int mm = m0 + v + hi;
        if (mm < M && nn < N)
            Cb[(long long)mm * ldc + nn] = acc[v] * scale;
    }
}

// ---------------------------------------------------------------------------
// Fused FFN:  Y = relu(X@W1 + b1) @ W2 + b2   (hidden tile kept in LDS)
//   grid = (ceil(M/16), 1, nbatch), block = 128.  d in {64,128}, f in {256,512}
//   (d, f multiples of 64 -> weight staging needs no guards).
// ---------------------------------------------------------------------------
__global__ void k_ffn(const float* __restrict__ X, long long bsX,
                      const float* __restrict__ W1, const float* __restrict__ b1,
                      const float* __restrict__ W2, const float* __restrict__ b2,
                      float* __restrict__ Y, long long bsY,
                      int M, int d, int f)
{
    __shared__ alignas(16) _Float16 Xh[16 * 128];
    __shared__ alignas(16) _Float16 Hh[16 * 512];
    __shared__ alignas(16) _Float16 Wt[64 * BT_LD];

    const int tid  = threadIdx.x;
    const int lane = tid & 31;
    const int wid  = tid >> 5;
    const int m0   = blockIdx.x * 16;
    const long long bz = blockIdx.z;
    const float* Xb = X + bz * bsX;

    // stage X tile (row-guard only; cols exact)
    for (int e = tid; e < 16 * d; e += 128) {
        int r = e / d, c = e % d;
        int row = m0 + r; if (row > M - 1) row = M - 1;
        float v = Xb[(long long)row * d + c];
        Xh[r * d + c] = (_Float16)((m0 + r < M) ? v : 0.f);
    }
    __syncthreads();

    // ---- stage 1: H = relu(X @ W1 + b1) -> LDS ----
    for (int n1 = 0; n1 < f; n1 += 64) {
        v8f acc = {};
        for (int k0 = 0; k0 < d; k0 += 64) {
            #pragma unroll
            for (int bb = 0; bb < 4; ++bb) {
                float wv[8];
                #pragma unroll
                for (int i = 0; i < 8; ++i) {
                    int e  = tid + 128 * (bb * 8 + i);
                    int kk = e >> 6, nn = e & 63;
                    wv[i] = W1[(long long)(k0 + kk) * f + n1 + nn];
                }
                #pragma unroll
                for (int i = 0; i < 8; ++i) {
                    int e  = tid + 128 * (bb * 8 + i);
                    int kk = e >> 6, nn = e & 63;
                    Wt[nn * BT_LD + kk] = (_Float16)wv[i];
                }
            }
            __syncthreads();
            #pragma unroll
            for (int c = 0; c < 2; ++c) {
                v16h a = load_afrag(Xh, d, lane, k0 + c * 32);
                v16h b = load_bfrag(Wt, lane, wid, c * 32);
                acc = __builtin_amdgcn_wmma_f32_16x16x32_f16(false, a, false, b,
                                                             (short)0, acc,
                                                             false, false);
            }
            __syncthreads();
        }
        const int hi = (lane >> 4) * 8, nn = n1 + 16 * wid + (lane & 15);
        #pragma unroll
        for (int v = 0; v < 8; ++v) {
            float h = acc[v] + b1[nn];
            Hh[(v + hi) * f + nn] = (_Float16)(h > 0.f ? h : 0.f);
        }
    }
    __syncthreads();

    // ---- stage 2: Y = H @ W2 + b2 ----
    float* Yb = Y + bz * bsY;
    for (int n2 = 0; n2 < d; n2 += 64) {
        v8f acc = {};
        for (int k0 = 0; k0 < f; k0 += 64) {
            #pragma unroll
            for (int bb = 0; bb < 4; ++bb) {
                float wv[8];
                #pragma unroll
                for (int i = 0; i < 8; ++i) {
                    int e  = tid + 128 * (bb * 8 + i);
                    int kk = e >> 6, nn = e & 63;
                    wv[i] = W2[(long long)(k0 + kk) * d + n2 + nn];
                }
                #pragma unroll
                for (int i = 0; i < 8; ++i) {
                    int e  = tid + 128 * (bb * 8 + i);
                    int kk = e >> 6, nn = e & 63;
                    Wt[nn * BT_LD + kk] = (_Float16)wv[i];
                }
            }
            __syncthreads();
            #pragma unroll
            for (int c = 0; c < 2; ++c) {
                v16h a = load_afrag(Hh, f, lane, k0 + c * 32);
                v16h b = load_bfrag(Wt, lane, wid, c * 32);
                acc = __builtin_amdgcn_wmma_f32_16x16x32_f16(false, a, false, b,
                                                             (short)0, acc,
                                                             false, false);
            }
            __syncthreads();
        }
        const int hi = (lane >> 4) * 8, nn = n2 + 16 * wid + (lane & 15);
        #pragma unroll
        for (int v = 0; v < 8; ++v) {
            int mm = m0 + v + hi;
            if (mm < M && nn < d)
                Yb[(long long)mm * d + nn] = acc[v] + b2[nn];
        }
    }
}

// ---------------------------------------------------------------------------
// Softmax (wave-per-row).  biasMode: 0=none, 1=bias[b, r, c], 2=bias[b, r%T, c]
// maskMode: 0=none, 1=keep iff (r<ds && c<ds), 2=keep iff (c<ds)
// ---------------------------------------------------------------------------
__global__ void k_softmax(float* __restrict__ S, int rowsPerBatch, int ncols,
                          int ldS, long long bsS,
                          const float* __restrict__ bias, int biasMode,
                          long long biasBS, int biasLd, int Tmod,
                          const int* __restrict__ dsz, int maskMode,
                          long long rowsTotal)
{
    const int lane = threadIdx.x & 31;
    const int wid  = threadIdx.x >> 5;
    long long rowIdx = (long long)blockIdx.x * 4 + wid;
    if (rowIdx >= rowsTotal) return;
    long long bz = rowIdx / rowsPerBatch;
    int r = (int)(rowIdx % rowsPerBatch);
    float* row = S + bz * bsS + (long long)r * ldS;
    int ds = maskMode ? dsz[bz] : 0;

    float vals[4];
    float mx = -3.0e38f;
    #pragma unroll
    for (int i = 0; i < 4; ++i) {
        int c = lane + 32 * i;
        float v = -3.0e38f;
        if (c < ncols) {
            v = row[c];
            if (biasMode == 1)      v += bias[bz * biasBS + (long long)r * biasLd + c];
            else if (biasMode == 2) v += bias[bz * biasBS + (long long)(r % Tmod) * biasLd + c];
            if (maskMode == 1)      { if (!(c < ds && r < ds)) v = -1.0e9f; }
            else if (maskMode == 2) { if (!(c < ds))           v = -1.0e9f; }
        }
        vals[i] = v;
        mx = fmaxf(mx, v);
    }
    for (int off = 16; off; off >>= 1) mx = fmaxf(mx, __shfl_xor(mx, off, 32));
    float sum = 0.f;
    #pragma unroll
    for (int i = 0; i < 4; ++i) {
        int c = lane + 32 * i;
        float e = (c < ncols) ? __expf(vals[i] - mx) : 0.f;
        vals[i] = e;
        sum += e;
    }
    for (int off = 16; off; off >>= 1) sum += __shfl_xor(sum, off, 32);
    float inv = 1.f / sum;
    #pragma unroll
    for (int i = 0; i < 4; ++i) {
        int c = lane + 32 * i;
        if (c < ncols) row[c] = vals[i] * inv;
    }
}

// ---------------------------------------------------------------------------
// Residual + LayerNorm (in place on X):  X = LN(X + O) * gs + gb
// ---------------------------------------------------------------------------
__global__ void k_ln(float* __restrict__ X, const float* __restrict__ O,
                     const float* __restrict__ gs, const float* __restrict__ gb,
                     int dim, long long rowsTotal)
{
    const int lane = threadIdx.x & 31;
    const int wid  = threadIdx.x >> 5;
    long long row = (long long)blockIdx.x * 4 + wid;
    if (row >= rowsTotal) return;
    float* x = X + row * dim;
    const float* o = O + row * dim;
    const int nPer = dim >> 5;  // 2 (d=64) or 4 (d=128)

    float vals[4];
    float sum = 0.f;
    for (int i = 0; i < nPer; ++i) {
        int c = lane + 32 * i;
        float v = x[c] + o[c];
        vals[i] = v;
        sum += v;
    }
    for (int off = 16; off; off >>= 1) sum += __shfl_xor(sum, off, 32);
    float mu = sum / (float)dim;
    float sq = 0.f;
    for (int i = 0; i < nPer; ++i) { float d = vals[i] - mu; sq += d * d; }
    for (int off = 16; off; off >>= 1) sq += __shfl_xor(sq, off, 32);
    float rs = rsqrtf(sq / (float)dim + 1e-5f);
    for (int i = 0; i < nPer; ++i) {
        int c = lane + 32 * i;
        x[c] = (vals[i] - mu) * rs * gs[c] + gb[c];
    }
}

// ---------------------------------------------------------------------------
// Small scalar kernels
// ---------------------------------------------------------------------------
__global__ void k_pos(const float* __restrict__ tm, const int* __restrict__ dsz,
                      float* __restrict__ pos)
{
    int b = threadIdx.x + blockIdx.x * blockDim.x;
    if (b >= Bc) return;
    const float* t = tm + (long long)b * Sc;
    float* p = pos + (long long)b * Sc;
    float sum = 0.f;
    for (int s = 0; s < Sc; ++s) {
        float sh = (s == 0) ? t[0] : t[s - 1];
        sum += (t[s] == 0.f) ? 0.f : (t[s] - sh);
    }
    float avg = sum / ((float)dsz[b] - 1.f);
    float c = 0.f;
    for (int s = 0; s < Sc; ++s) {
        float sh = (s == 0) ? t[0] : t[s - 1];
        float iv = (t[s] == 0.f) ? 0.f : (t[s] - sh);
        c += iv / avg;
        p[s] = (t[s] == 0.f) ? 0.f : (c + (float)(s + 1));
    }
}

__global__ void k_gather64(const int* __restrict__ idx, const float* __restrict__ emb,
                           float* __restrict__ out, long long n)
{
    long long i = (long long)blockIdx.x * 256 + threadIdx.x;
    if (i >= n * 64) return;
    long long r = i >> 6;
    int c = (int)(i & 63);
    out[i] = emb[(long long)idx[r] * 64 + c];
}

__global__ void k_concat_src(const int* __restrict__ src_loc,
                             const float* __restrict__ emb_loc,
                             const float* __restrict__ G,
                             const float* __restrict__ pos,
                             float* __restrict__ X)
{
    long long i = (long long)blockIdx.x * 256 + threadIdx.x;
    if (i >= (long long)Bc * Sc * Dc) return;
    long long t = i >> 7;
    int ch = (int)(i & 127);
    float v = (ch < 64) ? emb_loc[(long long)src_loc[t] * 64 + ch]
                        : G[t * 64 + ch - 64];
    float p   = pos[t];
    int half  = ch >> 1;
    float div = __powf(10000.f, (float)half * 2.f / 128.f);
    float ang = p / div;
    v += (ch & 1) ? __cosf(ang) : __sinf(ang);
    X[i] = v;
}

__global__ void k_str(const float* __restrict__ tm, const float* __restrict__ lat,
                      const float* __restrict__ lon, float* __restrict__ STR)
{
    long long i = (long long)blockIdx.x * 256 + threadIdx.x;
    if (i >= (long long)Bc * Sc * Sc) return;
    int c = (int)(i % Sc);
    long long tmp = i / Sc;
    int r = (int)(tmp % Sc);
    long long b = tmp / Sc;
    float la1 = lat[b * Sc + r] * RADc, la2 = lat[b * Sc + c] * RADc;
    float lo1 = lon[b * Sc + r] * RADc, lo2 = lon[b * Sc + c] * RADc;
    float sh = __sinf((la1 - la2) * 0.5f);
    float sl = __sinf((lo1 - lo2) * 0.5f);
    float h = sh * sh + __cosf(la1) * __cosf(la2) * sl * sl;
    h = fminf(fmaxf(h, 0.f), 1.f);
    float dist = 2.f * asinf(sqrtf(h)) * 6371.f;
    float tv = fabsf(tm[b * Sc + r] - tm[b * Sc + c]) / 86400.f;
    STR[i] = fminf(tv, 7.f) - tv + fminf(dist, 10.f) - dist;
}

__global__ void k_pm(const int* __restrict__ src_user, const float* __restrict__ emb_user,
                     const float* __restrict__ mw, const float* __restrict__ mb,
                     float* __restrict__ PM)
{
    int i = threadIdx.x + blockIdx.x * blockDim.x;  // over B*S
    if (i >= Bc * Sc) return;
    const float* e = emb_user + (long long)src_user[i] * 64;
    for (int t = 0; t < Tc; ++t) {
        float s = mb[t];
        for (int d = 0; d < 64; ++d) s += e[d] * mw[d * Tc + t];
        PM[(long long)i * Tc + t] = s;
    }
}

__global__ void k_pred(const float* __restrict__ STR, const float* __restrict__ PM,
                       float* __restrict__ PRED)
{
    int i = threadIdx.x + blockIdx.x * blockDim.x;  // over B*T*S
    if (i >= Bc * Tc * Sc) return;
    int ii = i % Sc;
    int t  = (i / Sc) % Tc;
    int b  = i / (Sc * Tc);
    float s = 0.f;
    for (int j = 0; j < Sc; ++j)
        s += STR[(long long)b * Sc * Sc + (long long)ii * Sc + j] *
             PM[((long long)b * Sc + j) * Tc + t];
    PRED[((long long)b * Tc + t) * Sc + ii] = s;
}

__global__ void k_mem(const float* __restrict__ IN, const float* __restrict__ mw,
                      const float* __restrict__ mb, float* __restrict__ OUT)
{
    int i = threadIdx.x + blockIdx.x * blockDim.x;  // over B*T*S
    if (i >= Bc * Tc * Sc) return;
    int ii = i % Sc;
    long long row = i / Sc;
    const float* p = IN + row * Sc;
    float s = mb[ii];
    for (int j = 0; j < Sc; ++j) s += p[j] * mw[(long long)j * Sc + ii];
    OUT[row * Sc + ii] = p[ii] + fmaxf(s, 0.f);
}

__global__ void k_concat_trg(const int* __restrict__ trg_loc,
                             const float* __restrict__ emb_loc,
                             const float* __restrict__ GY,
                             float* __restrict__ TRG, float* __restrict__ Y)
{
    long long i = (long long)blockIdx.x * 256 + threadIdx.x;
    if (i >= (long long)Bc * Rc * Dc) return;
    long long t = i >> 7;
    int ch = (int)(i & 127);
    float v = (ch < 64) ? emb_loc[(long long)trg_loc[t] * 64 + ch]
                        : GY[t * 64 + ch - 64];
    TRG[i] = v;
    Y[i]   = v;
}

__global__ void k_final(const float* __restrict__ Y, const float* __restrict__ TRG,
                        float* __restrict__ out, long long n)
{
    long long i = (long long)blockIdx.x * blockDim.x + threadIdx.x;
    if (i >= n) return;
    const float* y = Y + i * Dc;
    const float* t = TRG + i * Dc;
    float s = 0.f;
    for (int c = 0; c < Dc; ++c) s += y[c] * t[c];
    out[i] = s;
}

// ---------------------------------------------------------------------------
// Host orchestration
// ---------------------------------------------------------------------------
struct Layer {
    const float *wq, *wk, *wv, *w1, *b1, *w2, *b2, *l1s, *l1b, *l2s, *l2b;
};
static Layer get_layer(void* const* d_in, int base)
{
    Layer L;
    L.wq  = (const float*)d_in[base + 0];
    L.wk  = (const float*)d_in[base + 1];
    L.wv  = (const float*)d_in[base + 2];
    L.w1  = (const float*)d_in[base + 3];
    L.b1  = (const float*)d_in[base + 4];
    L.w2  = (const float*)d_in[base + 5];
    L.b2  = (const float*)d_in[base + 6];
    L.l1s = (const float*)d_in[base + 7];
    L.l1b = (const float*)d_in[base + 8];
    L.l2s = (const float*)d_in[base + 9];
    L.l2b = (const float*)d_in[base + 10];
    return L;
}

// One full self-attention transformer block operating in place on x.
static void self_block(hipStream_t st, float* x, int nb, int M, int d, int f,
                       const Layer& L,
                       float* Q, float* Kc, float* Vc, float* SCb, float* O,
                       const float* bias, int biasMode, long long biasBS,
                       int biasLd, int Tmod, const int* dsz, int maskMode)
{
    const long long bsX = (long long)M * d;
    dim3 blk(128);
    dim3 gq((M + 15) / 16, (d + 63) / 64, nb);
    k_gemm<<<gq, blk, 0, st>>>(x, bsX, d, L.wq, 0, d, 0, Q,  bsX, d, M, d, d, 1.f);
    k_gemm<<<gq, blk, 0, st>>>(x, bsX, d, L.wk, 0, d, 0, Kc, bsX, d, M, d, d, 1.f);
    k_gemm<<<gq, blk, 0, st>>>(x, bsX, d, L.wv, 0, d, 0, Vc, bsX, d, M, d, d, 1.f);

    const long long bsS = (long long)M * M;
    dim3 gsc((M + 15) / 16, (M + 63) / 64, nb);
    k_gemm<<<gsc, blk, 0, st>>>(Q, bsX, d, Kc, bsX, d, 1, SCb, bsS, M,
                                M, M, d, 1.f / sqrtf((float)d));

    const long long rowsTotal = (long long)nb * M;
    dim3 grow((unsigned)((rowsTotal + 3) / 4));
    k_softmax<<<grow, blk, 0, st>>>(SCb, M, M, M, bsS, bias, biasMode, biasBS,
                                    biasLd, Tmod, dsz, maskMode, rowsTotal);
    k_gemm<<<gq, blk, 0, st>>>(SCb, bsS, M, Vc, bsX, d, 0, O, bsX, d, M, d, M, 1.f);
    k_ln<<<grow, blk, 0, st>>>(x, O, L.l1s, L.l1b, d, rowsTotal);

    dim3 gf((M + 15) / 16, 1, nb);
    k_ffn<<<gf, blk, 0, st>>>(x, bsX, L.w1, L.b1, L.w2, L.b2, O, bsX, M, d, f);
    k_ln<<<grow, blk, 0, st>>>(x, O, L.l2s, L.l2b, d, rowsTotal);
}

extern "C" void kernel_launch(void* const* d_in, const int* in_sizes, int n_in,
                              void* d_out, int out_size, void* d_ws, size_t ws_size,
                              hipStream_t stream)
{
    (void)in_sizes; (void)n_in; (void)out_size; (void)ws_size;

    // inputs in setup_inputs() insertion order (params flattened recursively)
    const int*   src_user  = (const int*)  d_in[0];
    const int*   src_loc   = (const int*)  d_in[1];
    const int*   src_gps   = (const int*)  d_in[2];
    const float* src_time  = (const float*)d_in[3];
    const float* src_lat   = (const float*)d_in[4];
    const float* src_lon   = (const float*)d_in[5];
    const int*   trg_loc   = (const int*)  d_in[6];
    const int*   trg_gps   = (const int*)  d_in[7];
    const int*   data_size = (const int*)  d_in[8];
    const float* emb_user  = (const float*)d_in[9];
    const float* emb_loc   = (const float*)d_in[10];
    const float* emb_gps   = (const float*)d_in[11];
    Layer Lgeo = get_layer(d_in, 12);
    Layer Lenc = get_layer(d_in, 23);
    Layer Ldec = get_layer(d_in, 34);
    const float* merge_w = (const float*)d_in[45];
    const float* merge_b = (const float*)d_in[46];
    const float* mem_w   = (const float*)d_in[47];
    const float* mem_b   = (const float*)d_in[48];
    float* out = (float*)d_out;

    // ---- workspace carving (floats) ----
    float* W = (float*)d_ws;
    long long off = 0;
    auto alloc = [&](long long n) { float* p = W + off; off += n; return p; };
    float* X    = alloc((long long)Bc * Sc * Dc);
    float* G    = alloc((long long)Bc * Sc * DIMc);
    float* POS  = alloc((long long)Bc * Sc);
    float* STR  = alloc((long long)Bc * Sc * Sc);
    float* PM   = alloc((long long)Bc * Sc * Tc);
    float* PRED = alloc((long long)Bc * Tc * Sc);
    float* PRED2= alloc((long long)Bc * Tc * Sc);
    float* TRG  = alloc((long long)Bc * Rc * Dc);
    float* Y    = alloc((long long)Bc * Rc * Dc);
    float* GY   = alloc((long long)Bc * Rc * DIMc);
    float* KD   = alloc((long long)Bc * Sc * Dc);
    float* VD   = alloc((long long)Bc * Sc * Dc);
    float* Q    = alloc((long long)Bc * Rc * Dc);
    float* Kc_  = alloc((long long)Bc * Rc * DIMc);
    float* Vc_  = alloc((long long)Bc * Rc * DIMc);
    float* SCb  = alloc((long long)Bc * Rc * Sc);
    float* O    = alloc((long long)Bc * Rc * Dc);

    dim3 blk(128);

    // 1) time-aware positions
    k_pos<<<1, 64, 0, stream>>>(src_time, data_size, POS);

    // 2) src gps embeddings -> geo transformer (2 layers, d=64, seq=100)
    {
        long long n = (long long)Bc * Sc * 64;
        k_gather64<<<(unsigned)((n + 255) / 256), 256, 0, stream>>>(src_gps, emb_gps, G, (long long)Bc * Sc);
    }
    for (int l = 0; l < 2; ++l)
        self_block(stream, G, Bc, Sc, DIMc, 256, Lgeo, Q, Kc_, Vc_, SCb, O,
                   nullptr, 0, 0, 0, 1, nullptr, 0);

    // 3) src = concat(emb_loc, geo) + positional encoding
    {
        long long n = (long long)Bc * Sc * Dc;
        k_concat_src<<<(unsigned)((n + 255) / 256), 256, 0, stream>>>(src_loc, emb_loc, G, POS, X);
    }

    // 4) spatio-temporal relation matrix
    {
        long long n = (long long)Bc * Sc * Sc;
        k_str<<<(unsigned)((n + 255) / 256), 256, 0, stream>>>(src_time, src_lat, src_lon, STR);
    }

    // 5) encoder (2 layers, d=128, bias=str_mat, pad mask)
    for (int l = 0; l < 2; ++l)
        self_block(stream, X, Bc, Sc, Dc, 512, Lenc, Q, Kc_, Vc_, SCb, O,
                   STR, 1, (long long)Sc * Sc, Sc, 1, data_size, 1);

    // 6) pm -> pred -> 2 memory layers
    k_pm  <<<(Bc * Sc      + 127) / 128, 128, 0, stream>>>(src_user, emb_user, merge_w, merge_b, PM);
    k_pred<<<(Bc * Tc * Sc + 127) / 128, 128, 0, stream>>>(STR, PM, PRED);
    k_mem <<<(Bc * Tc * Sc + 127) / 128, 128, 0, stream>>>(PRED, mem_w, mem_b, PRED2);
    k_mem <<<(Bc * Tc * Sc + 127) / 128, 128, 0, stream>>>(PRED2, mem_w, mem_b, PRED);

    // 7) trg gps embeddings -> geo transformer (batch=B*NNEG, seq=T, d=64)
    {
        long long n = (long long)Bc * Rc * 64;
        k_gather64<<<(unsigned)((n + 255) / 256), 256, 0, stream>>>(trg_gps, emb_gps, GY, (long long)Bc * Rc);
    }
    for (int l = 0; l < 2; ++l)
        self_block(stream, GY, Bc * NNEGc, Tc, DIMc, 256, Lgeo, Q, Kc_, Vc_, SCb, O,
                   nullptr, 0, 0, 0, 1, nullptr, 0);
    {
        long long n = (long long)Bc * Rc * Dc;
        k_concat_trg<<<(unsigned)((n + 255) / 256), 256, 0, stream>>>(trg_loc, emb_loc, GY, TRG, Y);
    }

    // 8) decoder: precompute K,V from enc_out once (same params both layers)
    {
        dim3 g((Sc + 15) / 16, (Dc + 63) / 64, Bc);
        k_gemm<<<g, blk, 0, stream>>>(X, (long long)Sc * Dc, Dc, Ldec.wk, 0, Dc, 0,
                                      KD, (long long)Sc * Dc, Dc, Sc, Dc, Dc, 1.f);
        k_gemm<<<g, blk, 0, stream>>>(X, (long long)Sc * Dc, Dc, Ldec.wv, 0, Dc, 0,
                                      VD, (long long)Sc * Dc, Dc, Sc, Dc, Dc, 1.f);
    }
    for (int l = 0; l < 2; ++l) {
        dim3 gq((Rc + 15) / 16, (Dc + 63) / 64, Bc);
        k_gemm<<<gq, blk, 0, stream>>>(Y, (long long)Rc * Dc, Dc, Ldec.wq, 0, Dc, 0,
                                       Q, (long long)Rc * Dc, Dc, Rc, Dc, Dc, 1.f);
        dim3 gsc((Rc + 15) / 16, (Sc + 63) / 64, Bc);
        k_gemm<<<gsc, blk, 0, stream>>>(Q, (long long)Rc * Dc, Dc,
                                        KD, (long long)Sc * Dc, Dc, 1,
                                        SCb, (long long)Rc * Sc, Sc,
                                        Rc, Sc, Dc, 1.f / sqrtf((float)Dc));
        long long rowsTotal = (long long)Bc * Rc;
        dim3 grow((unsigned)((rowsTotal + 3) / 4));
        k_softmax<<<grow, blk, 0, stream>>>(SCb, Rc, Sc, Sc, (long long)Rc * Sc,
                                            PRED, 2, (long long)Tc * Sc, Sc, Tc,
                                            data_size, 2, rowsTotal);
        k_gemm<<<gq, blk, 0, stream>>>(SCb, (long long)Rc * Sc, Sc,
                                       VD, (long long)Sc * Dc, Dc, 0,
                                       O, (long long)Rc * Dc, Dc, Rc, Dc, Sc, 1.f);
        k_ln<<<grow, blk, 0, stream>>>(Y, O, Ldec.l1s, Ldec.l1b, Dc, rowsTotal);
        dim3 gf((Rc + 15) / 16, 1, Bc);
        k_ffn<<<gf, blk, 0, stream>>>(Y, (long long)Rc * Dc, Ldec.w1, Ldec.b1,
                                      Ldec.w2, Ldec.b2, O, (long long)Rc * Dc,
                                      Rc, Dc, 512);
        k_ln<<<grow, blk, 0, stream>>>(Y, O, Ldec.l2s, Ldec.l2b, Dc, rowsTotal);
    }

    // 9) out[b, nneg*T+t] = dot(y, trg)
    {
        long long n = (long long)Bc * Rc;
        k_final<<<(unsigned)((n + 127) / 128), 128, 0, stream>>>(Y, TRG, out, n);
    }
}
